// MultiHeadSelfAttention_1151051235605
// MI455X (gfx1250) — matrix operand order repaired and checked by reading, not compiled, not run on hardware
//
#include <hip/hip_runtime.h>
#include <hip/hip_bf16.h>
#include <math.h>

typedef __attribute__((ext_vector_type(8)))  _Float16 v8h;
typedef __attribute__((ext_vector_type(16))) _Float16 v16h;
typedef __attribute__((ext_vector_type(8)))  float    v8f;

#define NEGBIG (-1.0e30f)

// ---- WMMA fragment helpers (CDNA5 wave32 layouts, CDNA5 ISA) ----
// A 16x32 f16 (contiguous k order chosen here, B matches; not the ISA table order): lane m%16 holds row m; element i -> K = kb+i,
// kb = (lane<16 ? 0 : 16). Two contiguous 8-half (16B) loads.
__device__ __forceinline__ v16h load_a_frag(const _Float16* __restrict__ rowbase, int kbase) {
    v8h lo = *(const v8h*)(rowbase + kbase);
    v8h hi = *(const v8h*)(rowbase + kbase + 8);
    v16h r;
#pragma unroll
    for (int i = 0; i < 8; ++i) { r[i] = lo[i]; r[i + 8] = hi[i]; }
    return r;
}

__device__ __forceinline__ v8f wmma_f16(v16h a, v16h b, v8f c) {
    return __builtin_amdgcn_wmma_f32_16x16x32_f16(false, a, false, b, (short)0, c, false, false);
}

// ---------------------------------------------------------------------------
__global__ void cvt_f32_f16_kernel(const float* __restrict__ src,
                                   _Float16* __restrict__ dst, int n) {
    int stride = gridDim.x * blockDim.x;
    for (int i = blockIdx.x * blockDim.x + threadIdx.x; i < n; i += stride)
        dst[i] = (_Float16)src[i];
}

// ---------------------------------------------------------------------------
// QKV projection: qvk(4096 x 3072) = Xh(4096x1024) @ Wh(3072x1024)^T + in_b.
// One wave -> 32x64 output tile, software-pipelined K loop (prefetch next
// k-step's fragments while the current 8 WMMAs execute). Epilogue decodes
// channel m = d*48 + which*16 + h (reference reshape (B,T,D,3,H); which:
// 0=q, 1=v, 2=k) and scatters fp16 into Qh/Kh (b,h,t,d) and Vt (b,h,d,t).
__global__ void qkv_gemm_kernel(const _Float16* __restrict__ Xh,
                                const _Float16* __restrict__ Wh,
                                const float* __restrict__ in_b,
                                _Float16* __restrict__ Qh,
                                _Float16* __restrict__ Kh,
                                _Float16* __restrict__ Vt) {
    const int Cin = 1024, T = 2048, H = 16, D = 64;
    const int tilesN = 3072 / 64;  // 48
    int waveId = (blockIdx.x * blockDim.x + threadIdx.x) >> 5;
    int lane = threadIdx.x & 31;
    int l16 = lane & 15, half = lane >> 4;
    int tm = waveId / tilesN;      // 0..127 (32-row tiles)
    int tn = waveId % tilesN;
    int row0 = tm * 32;
    int col0 = tn * 64;

    v8f acc[2][4];
#pragma unroll
    for (int g = 0; g < 2; ++g)
#pragma unroll
        for (int s = 0; s < 4; ++s) acc[g][s] = (v8f){0,0,0,0,0,0,0,0};

    const _Float16* arow0 = Xh + (size_t)(row0 + l16) * Cin;
    const _Float16* arow1 = Xh + (size_t)(row0 + 16 + l16) * Cin;
    const int akoff = half * 16;
    const int bkoff = half * 16;

    v16h a0 = load_a_frag(arow0, akoff);
    v16h a1 = load_a_frag(arow1, akoff);
    v16h b[4];
#pragma unroll
    for (int s = 0; s < 4; ++s)
        b[s] = *(const v16h*)(Wh + (size_t)(col0 + s * 16 + l16) * Cin + bkoff);

    for (int k0 = 0; k0 < Cin; k0 += 32) {
        int kn = (k0 + 32) & (Cin - 1);  // branchless prefetch index (wraps once)
        v16h a0n = load_a_frag(arow0, kn + akoff);
        v16h a1n = load_a_frag(arow1, kn + akoff);
        v16h bn[4];
#pragma unroll
        for (int s = 0; s < 4; ++s)
            bn[s] = *(const v16h*)(Wh + (size_t)(col0 + s * 16 + l16) * Cin + kn + bkoff);
#pragma unroll
        for (int s = 0; s < 4; ++s) {
            acc[0][s] = wmma_f16(a0, b[s], acc[0][s]);
            acc[1][s] = wmma_f16(a1, b[s], acc[1][s]);
        }
        a0 = a0n; a1 = a1n;
#pragma unroll
        for (int s = 0; s < 4; ++s) b[s] = bn[s];
    }

#pragma unroll
    for (int s = 0; s < 4; ++s) {
        int m = col0 + s * 16 + l16;
        float bias = in_b[m];
        int h = m & 15;
        int which = (m >> 4) % 3;
        int d = m / 48;
#pragma unroll
        for (int g = 0; g < 2; ++g) {
#pragma unroll
            for (int r = 0; r < 8; ++r) {
                int n = row0 + g * 16 + r + half * 8;
                int bi = n / T, t = n % T;
                int bh = bi * H + h;
                _Float16 hv = (_Float16)(acc[g][s][r] + bias);
                if (which == 0)      Qh[((size_t)bh * T + t) * D + d] = hv;
                else if (which == 2) Kh[((size_t)bh * T + t) * D + d] = hv;
                else                 Vt[((size_t)bh * D + d) * T + t] = hv;
            }
        }
    }
}

// ---------------------------------------------------------------------------
// Flash attention: one wave per (b,h, 16-query tile). 32 keys per iteration,
// online softmax, O accumulated in fp32 (16x64). P staged via per-wave LDS to
// convert C-layout -> A-layout between the two WMMAs. V loads issue before the
// softmax (independent of it); next chunk's K fragments prefetch during PV.
__global__ void attn_kernel(const _Float16* __restrict__ Qh,
                            const _Float16* __restrict__ Kh,
                            const _Float16* __restrict__ Vt,
                            _Float16* __restrict__ Oh) {
    const int T = 2048, H = 16, D = 64, C = 1024;
    __shared__ _Float16 lds[8][16][32];
    int wslot = threadIdx.x >> 5;
    int waveId = (blockIdx.x * blockDim.x + threadIdx.x) >> 5;
    int lane = threadIdx.x & 31;
    int l16 = lane & 15, half = lane >> 4;
    const int qtilesPerBH = T / 16;  // 128
    int bh = waveId / qtilesPerBH;   // b*H + h
    int qt = waveId % qtilesPerBH;
    int i0 = qt * 16;

    const _Float16* Qbase = Qh + (size_t)bh * T * D;
    const _Float16* Kbase = Kh + (size_t)bh * T * D;
    const _Float16* Vbase = Vt + (size_t)bh * D * T;

    v16h qf0, qf1;
    {
        const _Float16* qrow = Qbase + (size_t)(i0 + l16) * D;
        qf0 = load_a_frag(qrow, 0 + half * 16);
        qf1 = load_a_frag(qrow, 32 + half * 16);
    }

    float mrow[8], lrow[8];
    v8f oacc[4];
#pragma unroll
    for (int r = 0; r < 8; ++r) { mrow[r] = NEGBIG; lrow[r] = 0.0f; }
#pragma unroll
    for (int dt = 0; dt < 4; ++dt) oacc[dt] = (v8f){0,0,0,0,0,0,0,0};

    const float scale = 0.125f;  // D^-0.5, D=64
    int numj = i0 + 16;          // causal bound (exclusive)

    // prime K fragments for jt = 0
    v16h kf[2][2];
#pragma unroll
    for (int sub = 0; sub < 2; ++sub) {
        const _Float16* krow = Kbase + (size_t)(sub * 16 + l16) * D;
        kf[sub][0] = *(const v16h*)(krow + half * 16);
        kf[sub][1] = *(const v16h*)(krow + 32 + half * 16);
    }

    for (int jt = 0; jt < numj; jt += 32) {
        v8f s[2];
#pragma unroll
        for (int sub = 0; sub < 2; ++sub) {
            s[sub] = (v8f){0,0,0,0,0,0,0,0};
            s[sub] = wmma_f16(qf0, kf[sub][0], s[sub]);
            s[sub] = wmma_f16(qf1, kf[sub][1], s[sub]);
        }
        // issue V loads now -- independent of softmax, consumed after LDS wait
        v16h vf[4];
#pragma unroll
        for (int dt = 0; dt < 4; ++dt) {
            const _Float16* vrow = Vbase + (size_t)(dt * 16 + l16) * T;
            vf[dt] = *(const v16h*)(vrow + jt + half * 16);
        }
        // scale + causal mask
#pragma unroll
        for (int sub = 0; sub < 2; ++sub) {
#pragma unroll
            for (int r = 0; r < 8; ++r) {
                int ig = i0 + r + half * 8;
                int jg = jt + sub * 16 + l16;
                float v = s[sub][r] * scale;
                s[sub][r] = (jg <= ig) ? v : NEGBIG;
            }
        }
        // online softmax over the 32 new columns; stage P into LDS
        float alpha[8];
#pragma unroll
        for (int r = 0; r < 8; ++r) {
            float mx = fmaxf(s[0][r], s[1][r]);
#pragma unroll
            for (int off = 1; off < 16; off <<= 1)
                mx = fmaxf(mx, __shfl_xor(mx, off, 16));
            float mnew = fmaxf(mrow[r], mx);
            float a = __expf(mrow[r] - mnew);
            float p0 = __expf(s[0][r] - mnew);
            float p1 = __expf(s[1][r] - mnew);
            float rs = p0 + p1;
#pragma unroll
            for (int off = 1; off < 16; off <<= 1)
                rs += __shfl_xor(rs, off, 16);
            lrow[r] = lrow[r] * a + rs;
            mrow[r] = mnew;
            alpha[r] = a;
            int Mrow = r + half * 8;
            lds[wslot][Mrow][l16]      = (_Float16)p0;
            lds[wslot][Mrow][16 + l16] = (_Float16)p1;
        }
        // rescale running O before accumulating new contribution
#pragma unroll
        for (int dt = 0; dt < 4; ++dt)
#pragma unroll
            for (int r = 0; r < 8; ++r) oacc[dt][r] *= alpha[r];

        // prefetch next chunk's K fragments (overlaps with PV WMMAs)
        int jn = (jt + 32 < numj) ? (jt + 32) : jt;  // branchless, stays in bounds
        v16h kfn[2][2];
#pragma unroll
        for (int sub = 0; sub < 2; ++sub) {
            const _Float16* krow = Kbase + (size_t)(jn + sub * 16 + l16) * D;
            kfn[sub][0] = *(const v16h*)(krow + half * 16);
            kfn[sub][1] = *(const v16h*)(krow + 32 + half * 16);
        }

        asm volatile("s_wait_dscnt 0" ::: "memory");  // intra-wave LDS RAW

        v16h pf = load_a_frag(&lds[wslot][l16][0], half * 16);
#pragma unroll
        for (int dt = 0; dt < 4; ++dt)
            oacc[dt] = wmma_f16(pf, vf[dt], oacc[dt]);

#pragma unroll
        for (int sub = 0; sub < 2; ++sub) {
            kf[sub][0] = kfn[sub][0];
            kf[sub][1] = kfn[sub][1];
        }
    }
    // finalize: O /= l, store fp16 into (b, t, h*64+d)
    int bi = bh / H, h = bh % H;
#pragma unroll
    for (int dt = 0; dt < 4; ++dt) {
#pragma unroll
        for (int r = 0; r < 8; ++r) {
            int ig = i0 + r + half * 8;
            float o = oacc[dt][r] / lrow[r];
            Oh[((size_t)bi * T + ig) * C + h * D + dt * 16 + l16] = (_Float16)o;
        }
    }
}

// ---------------------------------------------------------------------------
// Output projection: Y(4096x1024) = Oh @ OWh^T + out_b, fp32 result.
// 32x64 tile per wave, pipelined like qkv_gemm.
__global__ void out_gemm_kernel(const _Float16* __restrict__ Oh,
                                const _Float16* __restrict__ OWh,
                                const float* __restrict__ out_b,
                                float* __restrict__ Y) {
    const int Cin = 1024, Cout = 1024;
    const int tilesN = Cout / 64;  // 16
    int waveId = (blockIdx.x * blockDim.x + threadIdx.x) >> 5;
    int lane = threadIdx.x & 31;
    int l16 = lane & 15, half = lane >> 4;
    int tm = waveId / tilesN;      // 0..127
    int tn = waveId % tilesN;
    int row0 = tm * 32;
    int col0 = tn * 64;

    v8f acc[2][4];
#pragma unroll
    for (int g = 0; g < 2; ++g)
#pragma unroll
        for (int s = 0; s < 4; ++s) acc[g][s] = (v8f){0,0,0,0,0,0,0,0};

    const _Float16* arow0 = Oh + (size_t)(row0 + l16) * Cin;
    const _Float16* arow1 = Oh + (size_t)(row0 + 16 + l16) * Cin;
    const int akoff = half * 16;
    const int bkoff = half * 16;

    v16h a0 = load_a_frag(arow0, akoff);
    v16h a1 = load_a_frag(arow1, akoff);
    v16h b[4];
#pragma unroll
    for (int s = 0; s < 4; ++s)
        b[s] = *(const v16h*)(OWh + (size_t)(col0 + s * 16 + l16) * Cin + bkoff);

    for (int k0 = 0; k0 < Cin; k0 += 32) {
        int kn = (k0 + 32) & (Cin - 1);
        v16h a0n = load_a_frag(arow0, kn + akoff);
        v16h a1n = load_a_frag(arow1, kn + akoff);
        v16h bn[4];
#pragma unroll
        for (int s = 0; s < 4; ++s)
            bn[s] = *(const v16h*)(OWh + (size_t)(col0 + s * 16 + l16) * Cin + kn + bkoff);
#pragma unroll
        for (int s = 0; s < 4; ++s) {
            acc[0][s] = wmma_f16(a0, b[s], acc[0][s]);
            acc[1][s] = wmma_f16(a1, b[s], acc[1][s]);
        }
        a0 = a0n; a1 = a1n;
#pragma unroll
        for (int s = 0; s < 4; ++s) b[s] = bn[s];
    }

#pragma unroll
    for (int s = 0; s < 4; ++s) {
        int m = col0 + s * 16 + l16;
        float bias = out_b[m];
#pragma unroll
        for (int g = 0; g < 2; ++g) {
#pragma unroll
            for (int r = 0; r < 8; ++r) {
                int n = row0 + g * 16 + r + half * 8;
                Y[(size_t)n * Cout + m] = acc[g][s][r] + bias;
            }
        }
    }
}

// ---------------------------------------------------------------------------
extern "C" void kernel_launch(void* const* d_in, const int* in_sizes, int n_in,
                              void* d_out, int out_size, void* d_ws, size_t ws_size,
                              hipStream_t stream) {
    (void)in_sizes; (void)n_in; (void)out_size; (void)ws_size;
    const float* x     = (const float*)d_in[0];  // (2,2048,1024)
    const float* in_w  = (const float*)d_in[1];  // (3072,1024)
    const float* in_b  = (const float*)d_in[2];  // (3072,)
    const float* out_w = (const float*)d_in[3];  // (1024,1024)
    const float* out_b = (const float*)d_in[4];  // (1024,)
    float* y = (float*)d_out;

    char* ws = (char*)d_ws;
    const size_t NTOK = 4096, C = 1024, C3 = 3072;
    _Float16* Xh  = (_Float16*)(ws);                                   // 8 MB
    _Float16* Wh  = (_Float16*)(ws + 8u * 1024 * 1024);                // 6 MB
    _Float16* OWh = (_Float16*)(ws + 14u * 1024 * 1024);               // 2 MB
    _Float16* Qh  = (_Float16*)(ws + 16u * 1024 * 1024);               // 8 MB
    _Float16* Kh  = (_Float16*)(ws + 24u * 1024 * 1024);               // 8 MB
    _Float16* Vt  = (_Float16*)(ws + 32u * 1024 * 1024);               // 8 MB
    _Float16* Oh  = (_Float16*)(ws + 40u * 1024 * 1024);               // 8 MB

    cvt_f32_f16_kernel<<<2048, 256, 0, stream>>>(x, Xh, (int)(NTOK * C));
    cvt_f32_f16_kernel<<<2048, 256, 0, stream>>>(in_w, Wh, (int)(C3 * C));
    cvt_f32_f16_kernel<<<1024, 256, 0, stream>>>(out_w, OWh, (int)(C * C));

    // 128 row-tiles x 48 col-tiles = 6144 waves, 8 waves (256 thr) per block
    qkv_gemm_kernel<<<768, 256, 0, stream>>>(Xh, Wh, in_b, Qh, Kh, Vt);

    // 32 (b,h) x 128 q-tiles = 4096 waves
    attn_kernel<<<512, 256, 0, stream>>>(Qh, Kh, Vt, Oh);

    // 128 row-tiles x 16 col-tiles = 2048 waves
    out_gemm_kernel<<<256, 256, 0, stream>>>(Oh, OWh, out_b, y);
}
